// LocalSelfAttention_84670985273587
// MI455X (gfx1250) — matrix operand order repaired and checked by reading, not compiled, not run on hardware
//
#include <hip/hip_runtime.h>
#include <hip/hip_bf16.h>

// ---------------------------------------------------------------------------
// Problem constants (from reference): B=2, T=2048, C=1024, H=16, HD=64,
// WIN=33 (HALF=16, causal), spiral params RADIUS=6, OMEGA=1, KSPR=1, STEP=0.1
// ---------------------------------------------------------------------------
#define BB   2
#define TT   2048
#define CC   1024
#define HH   16
#define HDIM 64
#define HALO 16

#define SP_RADIUS 6.0f
#define SP_OMEGA  1.0f
#define SP_KSPR   1.0f
#define SP_STEP   0.1f
#define SP_EPS    1e-8f

// ---------------- bf16 / WMMA types ---------------------------------------
typedef __attribute__((ext_vector_type(16))) __bf16 v16bf;
typedef __attribute__((ext_vector_type(8)))  float  v8f;
typedef __attribute__((ext_vector_type(4)))  int    v4i;

__device__ __forceinline__ unsigned short f32_to_bf16u(float f) {
    union { float f; unsigned u; } v; v.f = f;
    unsigned lsb = (v.u >> 16) & 1u;
    unsigned r   = v.u + 0x7FFFu + lsb;   // round-to-nearest-even
    return (unsigned short)(r >> 16);
}

// ---------------- async global->LDS (gfx1250) ------------------------------
// Builtin's 1st param is v4i in AS1 (clang prints AS1 as `__device__`).
// AS conversions via integers: a generic global VA equals the AS1 address
// numerically, and the low 32 bits of a generic shared VA are the LDS byte
// offset (ISA 10.2 aperture rules) == AS3 pointer value.
#if defined(__gfx1250__) && __has_builtin(__builtin_amdgcn_global_load_async_to_lds_b128)
#define HAVE_ASYNC_LDS 1
typedef __attribute__((address_space(1))) v4i* as1_v4i_p;
typedef __attribute__((address_space(3))) v4i* as3_v4i_p;
__device__ __forceinline__ void async_copy16(const void* g, void* l) {
    __builtin_amdgcn_global_load_async_to_lds_b128(
        (as1_v4i_p)(unsigned long long)g,
        (as3_v4i_p)(unsigned)(unsigned long long)l,
        0, 0);
}
__device__ __forceinline__ void async_wait0() {
    asm volatile("s_wait_asynccnt 0" ::: "memory");
}
#endif

__global__ void cvt_f32_bf16_kernel(const float* __restrict__ in,
                                    unsigned short* __restrict__ out, int n) {
    int i = blockIdx.x * blockDim.x + threadIdx.x;
    if (i < n) out[i] = f32_to_bf16u(in[i]);
}

// ---------------- WMMA GEMM: C(f32) = A(bf16, MxK) * B(bf16, KxN) ----------
// Macro tile 128x64, BK=32, double-buffered LDS. 256 threads = 8 waves; wave
// w computes rows [w*16, w*16+16) of the macro tile across 64 cols (4 v8f).
// Fragments follow the wave32 layouts in CDNA5 ISA 7.12.2:
//   A: lane(m=L&15, half=L>>4), reg i: k = (i>>2)*16 + half*8 + (i&3)*2 (+0,1)
//   B: lane(n=L&15, half=L>>4), reg i: k = (i>>2)*16 + half*8 + (i&3)*2 (+0,1)  [B staged ^T]
// Row stride 40 ushorts (80 B) => every fragment is two 16B-aligned
// ds_load_b128's, and 16 consecutive rows hit distinct bank groups.
#define GBM 128
#define GBN 64
#define GBK 32
#define ASTR 40
#define BSTR 40

// Stage one K-tile: A via async global->LDS b128 copies, B via 4 global
// dword loads into registers followed by 8 transposing 16-bit DS stores.
__device__ __forceinline__ void stage_tiles(
    const unsigned short* __restrict__ A, const unsigned* __restrict__ Bg,
    unsigned short* AsBuf, unsigned short* BtBuf,
    int m0, int n0, int k0, int K, int Nd, int tid)
{
    // ---- A tile: 128 rows x 64 B ----
#ifdef HAVE_ASYNC_LDS
#pragma unroll
    for (int r = 0; r < 2; ++r) {
        int e   = tid + r * 256;
        int row = e >> 2;              // 4 16B chunks per row
        int c16 = e & 3;
        const char* g = (const char*)A + ((size_t)(m0 + row) * K + k0) * 2 + c16 * 16;
        char*       l = (char*)AsBuf + row * (ASTR * 2) + c16 * 16;
        async_copy16(g, l);
    }
#else
    const unsigned* Ag = (const unsigned*)A;
    const int Kd = K >> 1;
#pragma unroll
    for (int r = 0; r < 8; ++r) {
        int e   = tid + r * 256;
        int row = e >> 4;              // 16 dwords per A row
        int cd  = e & 15;
        unsigned d = Ag[(size_t)(m0 + row) * Kd + (k0 >> 1) + cd];
        *(unsigned*)&AsBuf[row * ASTR + cd * 2] = d;
    }
#endif
    // ---- B tile transposed: 32x64 ushort; regs first, then DS stores ----
    unsigned bd[4];
#pragma unroll
    for (int r = 0; r < 4; ++r) {
        int e   = tid + r * 256;
        int row = e >> 5;              // 32 dwords per B row
        int cd  = e & 31;
        bd[r] = Bg[(size_t)(k0 + row) * Nd + (n0 >> 1) + cd];
    }
#pragma unroll
    for (int r = 0; r < 4; ++r) {
        int e   = tid + r * 256;
        int row = e >> 5;
        int cd  = e & 31;
        int nl  = cd * 2;
        BtBuf[(nl + 0) * BSTR + row] = (unsigned short)(bd[r] & 0xFFFFu);
        BtBuf[(nl + 1) * BSTR + row] = (unsigned short)(bd[r] >> 16);
    }
}

template <int EPI>
__global__ __launch_bounds__(256)
void wmma_gemm_bf16_kernel(const unsigned short* __restrict__ A,
                           const unsigned short* __restrict__ Bm,
                           float* __restrict__ C,
                           int M, int N, int K,
                           const float* __restrict__ ln_scale,
                           const float* __restrict__ ln_delta,
                           const float* __restrict__ ln_bias) {
    __shared__ __align__(16) unsigned short As[2][GBM * ASTR];  // 20480 B
    __shared__ __align__(16) unsigned short Bt[2][GBN * BSTR];  // 10240 B

    const int tid  = threadIdx.x;
    const int wave = tid >> 5;
    const int lane = tid & 31;
    const int m0   = blockIdx.y * GBM;
    const int n0   = blockIdx.x * GBN;

    v8f acc[4];
#pragma unroll
    for (int i = 0; i < 4; ++i)
        acc[i] = (v8f){0.f, 0.f, 0.f, 0.f, 0.f, 0.f, 0.f, 0.f};

    const unsigned* Bg = (const unsigned*)Bm;
    const int Nd  = N >> 1;
    const int NKT = K / GBK;

    // pipeline prologue: stage tile 0 into buffer 0
    stage_tiles(A, Bg, As[0], Bt[0], m0, n0, 0, K, Nd, tid);

    for (int kt = 0; kt < NKT; ++kt) {
        const unsigned short* Asc = As[kt & 1];
        const unsigned short* Btc = Bt[kt & 1];
#ifdef HAVE_ASYNC_LDS
        async_wait0();                 // tile kt's async copies have landed
#endif
        __syncthreads();               // tile kt visible; prev reads of other buf done

        // stage tile kt+1 into the other buffer while computing tile kt
        if (kt + 1 < NKT)
            stage_tiles(A, Bg, As[(kt + 1) & 1], Bt[(kt + 1) & 1],
                        m0, n0, (kt + 1) * GBK, K, Nd, tid);

        // batch ALL fragment DS loads first: in-order DS returns let the
        // scheduler use graduated s_wait_dscnt thresholds so WMMA n overlaps
        // the loads of fragments n+1..3 instead of draining DScnt each time.
        union Frag { uint4 q[2]; v16bf v; };
        Frag af, bfr[4];
        const int mrow = wave * 16 + (lane & 15);
        const int hf   = lane >> 4;
        af.q[0] = *(const uint4*)&Asc[mrow * ASTR + hf * 8];
        af.q[1] = *(const uint4*)&Asc[mrow * ASTR + 16 + hf * 8];
#pragma unroll
        for (int nt = 0; nt < 4; ++nt) {
            const int ncol = nt * 16 + (lane & 15);
            bfr[nt].q[0] = *(const uint4*)&Btc[ncol * BSTR + hf * 8];
            bfr[nt].q[1] = *(const uint4*)&Btc[ncol * BSTR + 16 + hf * 8];
        }
#pragma unroll
        for (int nt = 0; nt < 4; ++nt)
            acc[nt] = __builtin_amdgcn_wmma_f32_16x16x32_bf16(
                false, af.v, false, bfr[nt].v, (short)0, acc[nt], false, false);
    }

    // epilogue (C/D layout: reg r, lanes0-15 -> M=r, lanes16-31 -> M=8+r)
    const int hf = lane >> 4;
    const int nn = lane & 15;
#pragma unroll
    for (int nt = 0; nt < 4; ++nt) {
#pragma unroll
        for (int r = 0; r < 8; ++r) {
            int m = m0 + wave * 16 + hf * 8 + r;
            int n = n0 + nt * 16 + nn;
            float v = acc[nt][r];
            if (EPI == 1)
                v = ln_delta[n] * tanhf(ln_scale[0] * v) + ln_bias[n];
            C[(size_t)m * N + n] = v;
        }
    }
}

// ---------------- spiral step (in place on fp32 qkv) -----------------------
__global__ void spiral_kernel(float* __restrict__ qkv, int total) {
    int idx = blockIdx.x * blockDim.x + threadIdx.x;
    if (idx >= total) return;
    int bt = idx / CC;
    int c  = idx - bt * CC;
    float* base = qkv + (size_t)bt * (3 * CC);
    float x0 = base[c];
    float x1 = base[CC + c];
    float x2 = base[2 * CC + c];
    const float cs = cosf(SP_OMEGA * SP_STEP);
    const float sn = sinf(SP_OMEGA * SP_STEP);
#pragma unroll
    for (int it = 0; it < 2; ++it) {
        float r = fmaxf(sqrtf(x0 * x0 + x1 * x1 + x2 * x2), SP_EPS);
        float coef = (SP_RADIUS - r) / r;
        float y0 = cs * x0 - sn * x1;
        float y1 = sn * x0 + cs * x1;
        float n0 = x0 + SP_STEP * (y0 - x0 + SP_KSPR * coef * x0);
        float n1 = x1 + SP_STEP * (y1 - x1 + SP_KSPR * coef * x1);
        float n2 = x2 + SP_STEP * (SP_KSPR * coef * x2);
        x0 = n0; x1 = n1; x2 = n2;
    }
    base[c]      = x0;   // q'
    base[CC + c] = x1;   // k'
    // v (base[2C+c]) intentionally left as the pre-spiral value (ref keeps it)
}

// ---------------- sliding-window causal attention with sink ----------------
// One thread per query; 64-query tile per block; K/V tile (+16 halo) in LDS.
#define TQ 64
__global__ __launch_bounds__(64)
void attn_kernel(const float* __restrict__ qkv,
                 const float* __restrict__ sinks,
                 unsigned short* __restrict__ ybf) {
    __shared__ __align__(16) float Ks[(TQ + HALO) * HDIM];  // 20 KB
    __shared__ __align__(16) float Vs[(TQ + HALO) * HDIM];  // 20 KB

    const int b  = blockIdx.z;
    const int h  = blockIdx.y;
    const int t0 = blockIdx.x * TQ;
    const int tid = threadIdx.x;
    const size_t C3 = (size_t)3 * CC;

    // cooperative K/V tile load: rows [t0-16, t0+TQ), 16B granular
    for (int e = tid; e < (TQ + HALO) * (HDIM / 4); e += TQ) {
        int rr = e >> 4;          // tile row
        int cq = e & 15;          // float4 within row
        int j  = t0 - HALO + rr;
        if (j >= 0) {             // j < TT guaranteed (j <= t0+TQ-1 < TT)
            const float* kp = qkv + (size_t)(b * TT + j) * C3 + CC + h * HDIM;
            const float* vp = qkv + (size_t)(b * TT + j) * C3 + 2 * CC + h * HDIM;
#ifdef HAVE_ASYNC_LDS
            async_copy16(kp + cq * 4, &Ks[e * 4]);
            async_copy16(vp + cq * 4, &Vs[e * 4]);
#else
            ((float4*)Ks)[e] = ((const float4*)kp)[cq];
            ((float4*)Vs)[e] = ((const float4*)vp)[cq];
#endif
        } else {
            float4 z = make_float4(0.f, 0.f, 0.f, 0.f);
            ((float4*)Ks)[e] = z;
            ((float4*)Vs)[e] = z;
        }
    }
#ifdef HAVE_ASYNC_LDS
    async_wait0();
#endif
    __syncthreads();

    const int t = t0 + tid;
    // load this query's q row
    float q[HDIM];
    const float* qp = qkv + (size_t)(b * TT + t) * C3 + h * HDIM;
#pragma unroll
    for (int d4 = 0; d4 < HDIM / 4; ++d4) {
        float4 f = ((const float4*)qp)[d4];
        q[d4 * 4 + 0] = f.x; q[d4 * 4 + 1] = f.y;
        q[d4 * 4 + 2] = f.z; q[d4 * 4 + 3] = f.w;
    }

    // scores over the 17 causal window positions j = t-16 .. t
    float sc[HALO + 1];
    float mx = -INFINITY;
#pragma unroll
    for (int w = 0; w <= HALO; ++w) {
        int j = t - HALO + w;
        float s = -INFINITY;
        if (j >= 0) {
            const float* kr = Ks + (tid + w) * HDIM;   // row = j-(t0-16) = tid+w
            float a = 0.f;
#pragma unroll
            for (int d = 0; d < HDIM; ++d) a += q[d] * kr[d];
            s = a * 0.125f;                            // 1/sqrt(64)
        }
        sc[w] = s;
        mx = fmaxf(mx, s);
    }
    const float sink = sinks[h];
    const float mj   = fmaxf(mx, sink);
    float denom = expf(sink - mj);

    float y[HDIM];
#pragma unroll
    for (int d = 0; d < HDIM; ++d) y[d] = 0.f;
#pragma unroll
    for (int w = 0; w <= HALO; ++w) {
        int j = t - HALO + w;
        if (j < 0) continue;
        float e = expf(sc[w] - mj);
        denom += e;
        const float* vr = Vs + (tid + w) * HDIM;
#pragma unroll
        for (int d = 0; d < HDIM; ++d) y[d] += e * vr[d];
    }
    const float inv = 1.f / denom;

    unsigned short* yp = ybf + (size_t)(b * TT + t) * CC + h * HDIM;
#pragma unroll
    for (int d = 0; d < HDIM; ++d) yp[d] = f32_to_bf16u(y[d] * inv);
}

// ---------------------------------------------------------------------------
extern "C" void kernel_launch(void* const* d_in, const int* in_sizes, int n_in,
                              void* d_out, int out_size, void* d_ws, size_t ws_size,
                              hipStream_t stream) {
    (void)in_sizes; (void)n_in; (void)out_size; (void)ws_size;

    const float* x        = (const float*)d_in[0];
    const float* W_attn   = (const float*)d_in[1];
    const float* W_proj   = (const float*)d_in[2];
    const float* sinks    = (const float*)d_in[3];
    const float* ln_scale = (const float*)d_in[4];
    const float* ln_delta = (const float*)d_in[5];
    const float* ln_bias  = (const float*)d_in[6];

    constexpr int M  = BB * TT;   // 4096
    constexpr int K  = CC;        // 1024
    constexpr int N1 = 3 * CC;    // 3072
    constexpr int N2 = CC;        // 1024

    // workspace layout (72 MB total)
    char* ws = (char*)d_ws;
    float*          qkv = (float*)ws;                               // 48 MB
    unsigned short* xbf = (unsigned short*)(ws + (size_t)M * N1 * 4);
    unsigned short* wab = xbf + (size_t)M * K;                      //  6 MB
    unsigned short* wpb = wab + (size_t)K * N1;                     //  2 MB
    unsigned short* ybf = wpb + (size_t)K * N2;                     //  8 MB

    // 1) fp32 -> bf16 converts
    cvt_f32_bf16_kernel<<<(M * K + 255) / 256, 256, 0, stream>>>(x, xbf, M * K);
    cvt_f32_bf16_kernel<<<(K * N1 + 255) / 256, 256, 0, stream>>>(W_attn, wab, K * N1);
    cvt_f32_bf16_kernel<<<(K * N2 + 255) / 256, 256, 0, stream>>>(W_proj, wpb, K * N2);

    // 2) qkv = x @ W_attn   (bf16 WMMA, fp32 accumulate)
    dim3 g1(N1 / GBN, M / GBM);
    wmma_gemm_bf16_kernel<0><<<g1, 256, 0, stream>>>(
        xbf, wab, qkv, M, N1, K, nullptr, nullptr, nullptr);

    // 3) spiral step x2, in place on qkv (q,k updated; v kept)
    spiral_kernel<<<(M * CC + 255) / 256, 256, 0, stream>>>(qkv, M * CC);

    // 4) windowed causal attention with sink -> ybf (bf16)
    dim3 g2(TT / TQ, HH, BB);
    attn_kernel<<<g2, TQ, 0, stream>>>(qkv, sinks, ybf);

    // 5) out = ln_delta * tanh(ln_scale * (y @ W_proj)) + ln_bias
    dim3 g3(N2 / GBN, M / GBM);
    wmma_gemm_bf16_kernel<1><<<g3, 256, 0, stream>>>(
        ybf, wpb, (float*)d_out, M, N2, K, ln_scale, ln_delta, ln_bias);
}
